// AttentionRNN_65292092834059
// MI455X (gfx1250) — matrix operand order repaired and checked by reading, compiled, not run on hardware
//
#include <hip/hip_runtime.h>

// ---------------------------------------------------------------------------
// Problem constants (match reference)
// ---------------------------------------------------------------------------
static constexpr int Bn = 128;   // batch
static constexpr int Sn = 512;   // seq len
static constexpr int Vn = 128;   // vocab
static constexpr int En = 64;    // embed
static constexpr int Hn = 256;   // hidden

// ---------------------------------------------------------------------------
// WMMA types / helpers (CDNA5 wave32, 16x16x32 bf16 -> f32)
// ---------------------------------------------------------------------------
typedef __attribute__((ext_vector_type(16))) __bf16 v16bf;
typedef __attribute__((ext_vector_type(8)))  float  v8f;

__device__ __forceinline__ v8f wmma_bf16(v16bf a, v16bf b, v8f c) {
  // (neg_a, A, neg_b, B, c_mod, C, reuse_a, reuse_b)
  return __builtin_amdgcn_wmma_f32_16x16x32_bf16(false, a, false, b, (short)0, c, false, false);
}

// A fragment (16x32 bf16) from row-major LDS tile, per ISA layout:
// lanes 0-15: row M=lane, K = {kc..kc+7, kc+16..kc+23}
// lanes 16-31: row M=lane-16, K = {kc+8..kc+15, kc+24..kc+31}
__device__ __forceinline__ v16bf load_a_lds(const __bf16* A, int lda, int kc, int lane) {
  const int m  = lane & 15;
  const int k8 = (lane >> 4) << 3;   // 0 or 8
  const __bf16* row = A + m * lda + kc + k8;
  v16bf a;
#pragma unroll
  for (int e = 0; e < 8; ++e) { a[e] = row[e]; a[e + 8] = row[16 + e]; }
  return a;
}

// B fragment (32x16 bf16) from pre-swizzled fragment storage:
// frag[((kc32*NT + nt)*32 + lane)*16 + e]  — one contiguous 32B load per lane.
__device__ __forceinline__ v16bf load_b_frag(const __bf16* frag, int NT, int kc32, int nt, int lane) {
  return *reinterpret_cast<const v16bf*>(frag + ((((size_t)kc32 * NT + nt) * 32 + lane) << 4));
}

__device__ __forceinline__ float sigm(float x) { return 1.0f / (1.0f + expf(-x)); }

// bf16 pair unpack from a packed uint32 (bf16 -> f32 is a 16-bit shift)
__device__ __forceinline__ float bf_lo(unsigned w) { return __uint_as_float(w << 16); }
__device__ __forceinline__ float bf_hi(unsigned w) { return __uint_as_float(w & 0xffff0000u); }

// Async global -> LDS copy, 16 bytes per lane (tracked on ASYNCcnt).
// ldsOff is the LDS byte offset (low 32 bits of a __shared__ pointer).
__device__ __forceinline__ void async_g2l_b128(unsigned ldsOff, const void* gptr) {
  asm volatile("global_load_async_to_lds_b128 %0, %1, off"
               :: "v"(ldsOff), "v"(gptr)
               : "memory");
}
__device__ __forceinline__ void async_wait0() {
  asm volatile("s_wait_asynccnt 0" ::: "memory");
}
__device__ __forceinline__ unsigned lds_off(const void* p) {
  return (unsigned)(unsigned long long)p;   // LDS aperture: addr[31:0] = offset
}

// ---------------------------------------------------------------------------
// Prologue: pack a logical B-matrix B[k][n] = src[n][k] (row-major src, f32)
// into bf16 WMMA fragment order. splitN lets us concatenate two sources
// ([W_h | W_s]) along N.
// ---------------------------------------------------------------------------
__global__ __launch_bounds__(256)
void k_pack_b(const float* __restrict__ srcA, const float* __restrict__ srcB,
              int splitN, int N, int srcStride, __bf16* __restrict__ dst, int total) {
  int idx = blockIdx.x * 256 + threadIdx.x;
  if (idx >= total) return;
  const int e    = idx & 15;
  const int lane = (idx >> 4) & 31;
  const int tile = idx >> 9;
  const int NT   = N >> 4;
  const int nt   = tile % NT;
  const int kc   = tile / NT;
  const int n = nt * 16 + (lane & 15);
  const int k = kc * 32 + ((lane >> 4) << 3) + (e & 7) + ((e >> 3) << 4);
  float v = (n < splitN) ? srcA[n * srcStride + k] : srcB[(n - splitN) * srcStride + k];
  dst[idx] = (__bf16)v;
}

__global__ __launch_bounds__(256)
void k_f2bf(const float* __restrict__ src, __bf16* __restrict__ dst, int n) {
  int i = blockIdx.x * 256 + threadIdx.x;
  if (i < n) dst[i] = (__bf16)src[i];
}

// ---------------------------------------------------------------------------
// K1: gates GEMM + LSTM cell (in registers) + [W_h|W_s] projection GEMM.
// grid = B/16 workgroups, 256 threads (8 waves). Each wave owns 2 h-blocks,
// computing i/f/g/o tiles together so the cell update needs no gate staging.
// ---------------------------------------------------------------------------
__global__ __launch_bounds__(256)
void k_lstm_step(const int* __restrict__ x, const __bf16* __restrict__ embBf,
                 const __bf16* __restrict__ wgFrag, const float* __restrict__ b_gates,
                 const __bf16* __restrict__ whwsFrag,
                 __bf16* __restrict__ hBf, float* __restrict__ cF,
                 __bf16* __restrict__ HbufBf, __bf16* __restrict__ WhHBf,
                 float* __restrict__ wshF, int t) {
  const int b0   = blockIdx.x * 16;
  const int tid  = threadIdx.x;
  const int wave = tid >> 5;
  const int lane = tid & 31;

  __shared__ __bf16 Xcat[16 * 320];   // [emb(x_t) | h_prev], 16 rows
  __shared__ __bf16 Hnew[16 * 256];   // fresh h, bf16

  // h-part: straight global->LDS copy via CDNA5 async path (16B/lane, 2 iters)
  for (int i = tid; i < 16 * 32; i += 256) {         // 16 rows x 32 chunks of 8 bf16
    const int r = i >> 5, j = i & 31;
    async_g2l_b128(lds_off(&Xcat[r * 320 + En + j * 8]),
                   hBf + (b0 + r) * Hn + j * 8);
  }
  // emb-part: token-dependent gather (normal path)
  for (int i = tid; i < 16 * En; i += 256) {
    const int r = i >> 6, k = i & 63;
    const int tok = x[(b0 + r) * Sn + t];
    Xcat[r * 320 + k] = embBf[tok * En + k];
  }
  async_wait0();
  __syncthreads();

  const int col    = lane & 15;
  const int rowOff = (lane >> 4) << 3;   // C/D: M = r + 8*(lane>=16)

  // ---- gates GEMM: [16,320] x [320,1024], i/f/g/o tiles per h-block ----
  for (int hb = wave * 2; hb < wave * 2 + 2; ++hb) {
    v8f ai = {}, af = {}, ag = {}, ao = {};
#pragma unroll
    for (int kc32 = 0; kc32 < 10; ++kc32) {           // K = 320
      v16bf a = load_a_lds(Xcat, 320, kc32 * 32, lane);
      ai = wmma_bf16(a, load_b_frag(wgFrag, 64, kc32, hb,      lane), ai);
      af = wmma_bf16(a, load_b_frag(wgFrag, 64, kc32, hb + 16, lane), af);
      ag = wmma_bf16(a, load_b_frag(wgFrag, 64, kc32, hb + 32, lane), ag);
      ao = wmma_bf16(a, load_b_frag(wgFrag, 64, kc32, hb + 48, lane), ao);
    }
    const int hidx = hb * 16 + col;
    const float bi = b_gates[hidx],        bf = b_gates[256 + hidx];
    const float bg = b_gates[512 + hidx],  bo = b_gates[768 + hidx];
#pragma unroll
    for (int r = 0; r < 8; ++r) {
      const int row  = r + rowOff;
      const int bidx = (b0 + row) * Hn + hidx;
      float gi = sigm(ai[r] + bi);
      float gf = sigm(af[r] + bf);
      float gg = tanhf(ag[r] + bg);
      float go = sigm(ao[r] + bo);
      float cn = gf * cF[bidx] + gi * gg;
      float hn = go * tanhf(cn);
      cF[bidx] = cn;
      __bf16 h16 = (__bf16)hn;
      hBf[bidx] = h16;
      HbufBf[((size_t)(b0 + row) * Sn + t) * Hn + hidx] = h16;
      Hnew[row * 256 + hidx] = h16;
    }
  }
  __syncthreads();

  // ---- projection GEMM: [16,256] x [256,512] = [h@W_h.T | h@W_s.T] ----
  for (int ct = wave * 4; ct < wave * 4 + 4; ++ct) {
    v8f acc = {};
#pragma unroll
    for (int kc32 = 0; kc32 < 8; ++kc32) {            // K = 256
      v16bf a = load_a_lds(Hnew, 256, kc32 * 32, lane);
      acc = wmma_bf16(a, load_b_frag(whwsFrag, 32, kc32, ct, lane), acc);
    }
    const int n = ct * 16 + col;
#pragma unroll
    for (int r = 0; r < 8; ++r) {
      const int row = r + rowOff;
      if (n < 256)
        WhHBf[((size_t)(b0 + row) * Sn + t) * Hn + n] = (__bf16)acc[r];
      else
        wshF[(b0 + row) * Hn + (n - 256)] = acc[r];
    }
  }
}

// ---------------------------------------------------------------------------
// K2: Bahdanau attention for one batch row per workgroup. f32 softmax.
// Bandwidth-critical: 16B/lane vectorized bf16 reads + bit-op unpacks.
// ---------------------------------------------------------------------------
__global__ __launch_bounds__(256)
void k_attn(const int* __restrict__ lengths, const float* __restrict__ v_attn,
            const __bf16* __restrict__ WhHBf, const float* __restrict__ wshF,
            const __bf16* __restrict__ HbufBf, __bf16* __restrict__ ctxBf, int t) {
  const int b   = blockIdx.x;
  const int tid = threadIdx.x;
  if (t == 0) {                 // nothing to attend to: ctx = 0
    ctxBf[b * Hn + tid] = (__bf16)0.0f;
    return;
  }
  __shared__ float sc[Sn];
  __shared__ float red[256];
  __shared__ float vsh[Hn];
  __shared__ float wsh[Hn];
  __shared__ float ctxs[2][Hn];

  const int len  = lengths[b];
  const int tEff = (t < len) ? t : len;   // len >= 1, t >= 1  ->  tEff >= 1
  vsh[tid] = v_attn[tid];
  wsh[tid] = wshF[b * Hn + tid];
  __syncthreads();

  // scores: one s per thread iteration; row read as 32 x uint4 (16B chunks)
  for (int s = tid; s < tEff; s += 256) {
    const uint4* wrow = reinterpret_cast<const uint4*>(WhHBf + ((size_t)b * Sn + s) * Hn);
    float acc = 0.0f;
    for (int q = 0; q < Hn / 8; ++q) {
      const uint4 w = wrow[q];
      const int h = q * 8;
      acc += vsh[h + 0] * tanhf(bf_lo(w.x) + wsh[h + 0]);
      acc += vsh[h + 1] * tanhf(bf_hi(w.x) + wsh[h + 1]);
      acc += vsh[h + 2] * tanhf(bf_lo(w.y) + wsh[h + 2]);
      acc += vsh[h + 3] * tanhf(bf_hi(w.y) + wsh[h + 3]);
      acc += vsh[h + 4] * tanhf(bf_lo(w.z) + wsh[h + 4]);
      acc += vsh[h + 5] * tanhf(bf_hi(w.z) + wsh[h + 5]);
      acc += vsh[h + 6] * tanhf(bf_lo(w.w) + wsh[h + 6]);
      acc += vsh[h + 7] * tanhf(bf_hi(w.w) + wsh[h + 7]);
    }
    sc[s] = acc;
  }
  __syncthreads();

  float m = -3.0e38f;
  for (int s = tid; s < tEff; s += 256) m = fmaxf(m, sc[s]);
  red[tid] = m; __syncthreads();
  for (int off = 128; off > 0; off >>= 1) {
    if (tid < off) red[tid] = fmaxf(red[tid], red[tid + off]);
    __syncthreads();
  }
  const float mx = red[0];
  __syncthreads();

  float sum = 0.0f;
  for (int s = tid; s < tEff; s += 256) { float e = expf(sc[s] - mx); sc[s] = e; sum += e; }
  red[tid] = sum; __syncthreads();
  for (int off = 128; off > 0; off >>= 1) {
    if (tid < off) red[tid] += red[tid + off];
    __syncthreads();
  }
  const float inv = 1.0f / red[0];

  // ctx: thread owns an h-pair (packed u32 loads); s-range split over 2 groups
  const int g = tid >> 7;          // 0 / 1
  const int p = tid & 127;         // h-pair index
  float a0 = 0.0f, a1 = 0.0f;
  for (int s = g; s < tEff; s += 2) {
    const unsigned w2 = *reinterpret_cast<const unsigned*>(
        HbufBf + ((size_t)b * Sn + s) * Hn + 2 * p);
    const float wgt = sc[s];
    a0 += wgt * bf_lo(w2);
    a1 += wgt * bf_hi(w2);
  }
  ctxs[g][2 * p]     = a0;
  ctxs[g][2 * p + 1] = a1;
  __syncthreads();
  ctxBf[b * Hn + tid] = (__bf16)((ctxs[0][tid] + ctxs[1][tid]) * inv);
}

// ---------------------------------------------------------------------------
// K3: combined = tanh([h|ctx] @ W_comb.T + b), logits = combined @ W_out.T + b
// ---------------------------------------------------------------------------
__global__ __launch_bounds__(256)
void k_out(const __bf16* __restrict__ hBf, const __bf16* __restrict__ ctxBf,
           const __bf16* __restrict__ wcFrag, const float* __restrict__ b_comb,
           const __bf16* __restrict__ woFrag, const float* __restrict__ b_out,
           float* __restrict__ out, int t) {
  const int b0   = blockIdx.x * 16;
  const int tid  = threadIdx.x;
  const int wave = tid >> 5;
  const int lane = tid & 31;

  __shared__ __bf16 Xc[16 * 512];     // [h | ctx]
  __shared__ __bf16 Comb[16 * 256];

  // stage [h | ctx] via CDNA5 async global->LDS (16B/lane, 4 iters)
  for (int i = tid; i < 16 * 64; i += 256) {   // 16 rows x 64 chunks of 8 bf16
    const int r = i >> 6, j = i & 63;
    const __bf16* src = (j < 32) ? (hBf   + (b0 + r) * Hn + j * 8)
                                 : (ctxBf + (b0 + r) * Hn + (j - 32) * 8);
    async_g2l_b128(lds_off(&Xc[r * 512 + j * 8]), src);
  }
  async_wait0();
  __syncthreads();

  const int col    = lane & 15;
  const int rowOff = (lane >> 4) << 3;

  for (int ct = wave * 2; ct < wave * 2 + 2; ++ct) {   // N = 256 -> 16 tiles
    v8f acc = {};
#pragma unroll
    for (int kc32 = 0; kc32 < 16; ++kc32) {            // K = 512
      v16bf a = load_a_lds(Xc, 512, kc32 * 32, lane);
      acc = wmma_bf16(a, load_b_frag(wcFrag, 16, kc32, ct, lane), acc);
    }
    const int n  = ct * 16 + col;
    const float bc = b_comb[n];
#pragma unroll
    for (int r = 0; r < 8; ++r)
      Comb[(r + rowOff) * 256 + n] = (__bf16)tanhf(acc[r] + bc);
  }
  __syncthreads();

  {                                                    // N = 128 -> 8 tiles
    const int ct = wave;
    v8f acc = {};
#pragma unroll
    for (int kc32 = 0; kc32 < 8; ++kc32) {             // K = 256
      v16bf a = load_a_lds(Comb, 256, kc32 * 32, lane);
      acc = wmma_bf16(a, load_b_frag(woFrag, 8, kc32, ct, lane), acc);
    }
    const int n  = ct * 16 + col;
    const float bo = b_out[n];
#pragma unroll
    for (int r = 0; r < 8; ++r) {
      const int row = r + rowOff;
      out[((size_t)(b0 + row) * Sn + t) * Vn + n] = acc[r] + bo;
    }
  }
}

// ---------------------------------------------------------------------------
// Host: prologue (weight swizzle to bf16 fragments, state init) + 512-step loop
// ---------------------------------------------------------------------------
extern "C" void kernel_launch(void* const* d_in, const int* in_sizes, int n_in,
                              void* d_out, int out_size, void* d_ws, size_t ws_size,
                              hipStream_t stream) {
  (void)in_sizes; (void)n_in; (void)out_size; (void)ws_size;
  const int*   x   = (const int*)  d_in[0];
  const int*   len = (const int*)  d_in[1];
  const float* emb = (const float*)d_in[2];
  const float* Wg  = (const float*)d_in[3];
  const float* bg  = (const float*)d_in[4];
  const float* Wh  = (const float*)d_in[5];
  const float* Ws  = (const float*)d_in[6];
  const float* vat = (const float*)d_in[7];
  const float* Wc  = (const float*)d_in[8];
  const float* bc  = (const float*)d_in[9];
  const float* Wo  = (const float*)d_in[10];
  const float* bo  = (const float*)d_in[11];
  float* out = (float*)d_out;

  char* ws = (char*)d_ws;
  size_t off = 0;
  auto take = [&](size_t bytes) -> char* {
    char* p = ws + off;
    off = (off + bytes + 255) & ~(size_t)255;
    return p;
  };
  __bf16* wgFrag   = (__bf16*)take((size_t)320 * 1024 * 2);
  __bf16* whwsFrag = (__bf16*)take((size_t)256 * 512 * 2);
  __bf16* wcFrag   = (__bf16*)take((size_t)512 * 256 * 2);
  __bf16* woFrag   = (__bf16*)take((size_t)256 * 128 * 2);
  __bf16* embBf    = (__bf16*)take((size_t)Vn * En * 2);
  __bf16* hBf      = (__bf16*)take((size_t)Bn * Hn * 2);
  __bf16* ctxBf    = (__bf16*)take((size_t)Bn * Hn * 2);
  float*  cF       = (float*) take((size_t)Bn * Hn * 4);
  float*  wshF     = (float*) take((size_t)Bn * Hn * 4);
  __bf16* HbufBf   = (__bf16*)take((size_t)Bn * Sn * Hn * 2);
  __bf16* WhHBf    = (__bf16*)take((size_t)Bn * Sn * Hn * 2);

  // --- prologue: swizzle weights into WMMA B-fragment order (bf16) ---
  int total;
  total = 320 * 1024;   // gates:  B[k][n] = W_gates[n][k]
  k_pack_b<<<(total + 255) / 256, 256, 0, stream>>>(Wg, Wg, 1024, 1024, 320, wgFrag, total);
  total = 256 * 512;    // [W_h | W_s] transposed
  k_pack_b<<<(total + 255) / 256, 256, 0, stream>>>(Wh, Ws, 256, 512, 256, whwsFrag, total);
  total = 512 * 256;    // W_comb transposed
  k_pack_b<<<(total + 255) / 256, 256, 0, stream>>>(Wc, Wc, 256, 256, 512, wcFrag, total);
  total = 256 * 128;    // W_out transposed
  k_pack_b<<<(total + 255) / 256, 256, 0, stream>>>(Wo, Wo, 128, 128, 256, woFrag, total);
  k_f2bf<<<(Vn * En + 255) / 256, 256, 0, stream>>>(emb, embBf, Vn * En);
  (void)hipMemsetAsync(hBf, 0, (size_t)Bn * Hn * 2, stream);
  (void)hipMemsetAsync(cF,  0, (size_t)Bn * Hn * 4, stream);

  // --- sequential recurrence ---
  for (int t = 0; t < Sn; ++t) {
    k_lstm_step<<<Bn / 16, 256, 0, stream>>>(x, embBf, wgFrag, bg, whwsFrag,
                                             hBf, cF, HbufBf, WhHBf, wshF, t);
    k_attn<<<Bn, 256, 0, stream>>>(len, vat, WhHBf, wshF, HbufBf, ctxBf, t);
    k_out<<<Bn / 16, 256, 0, stream>>>(hBf, ctxBf, wcFrag, bc, woFrag, bo, out, t);
  }
}
